// TransformerWithRelativeAttention_16269336117997
// MI455X (gfx1250) — matrix operand order repaired and checked by reading, not compiled, not run on hardware
//
#include <hip/hip_runtime.h>
#include <hip/hip_bf16.h>
#include <cstdint>
#include <cstddef>

typedef _Float16 h16;
typedef __attribute__((ext_vector_type(8)))  _Float16 v8h;
typedef __attribute__((ext_vector_type(16))) _Float16 v16h;
typedef __attribute__((ext_vector_type(8)))  float    v8f;

// ---------------- problem constants ----------------
#define KB   8
#define KC   384
#define KL   1024          // 32*32 tokens per image
#define KT   (KB * KL)     // 8192 tokens
#define KNH  8
#define KDK  48
#define KDKP 64            // head dim padded to 64 (zeros) for WMMA K
#define KDV  48
#define KCFF 1536
#define KNB  3969          // (2*32-1)^2

// LDS B-tile: 64 weight rows x 32 K-halfs, row padded 32->40 halfs (80B) so the
// 16 lanes of a fragment read hit 16 disjoint bank quads (20 banks/row, gcd(20,64)=4)
#define BROW 40
#define BUFH (64 * BROW)   // halfs per buffer (double-buffered)

// ---------------- WMMA helpers ----------------
__device__ __forceinline__ v16h joinv(v8h lo, v8h hi) {
  return __builtin_shufflevector(lo, hi, 0,1,2,3,4,5,6,7,8,9,10,11,12,13,14,15);
}

__device__ __forceinline__ v8f wmma_f16(v16h a, v16h b, v8f c) {
  // D = A(16x32 f16) * B(32x16 f16) + C(16x16 f32)
  return __builtin_amdgcn_wmma_f32_16x16x32_f16(false, a, false, b, (short)0, c, false, false);
}

// A fragment: rowptr = &A[(m0 + (lane&15)) * lda]; lanes<16 hold K {0..7,16..23}, lanes>=16 {8..15,24..31}
__device__ __forceinline__ v16h load_af(const h16* __restrict__ rowptr, int ks, int lane) {
  const int ka = (lane >> 4) * 8;
  v8h lo = *(const v8h*)(rowptr + ks + ka);
  v8h hi = *(const v8h*)(rowptr + ks + 16 + ka);
  return joinv(lo, hi);
}

// B fragment: memory holds B transposed (row n is K-contiguous). rowptr = &W[(n0+(lane&15))*ldw]
__device__ __forceinline__ v16h load_bf(const h16* __restrict__ rowptr, int ks, int lane) {
  const int kb = (lane >> 4) * 8;
  v8h lo = *(const v8h*)(rowptr + ks + kb);
  v8h hi = *(const v8h*)(rowptr + ks + kb + 16);
  return joinv(lo, hi);
}

// ---------------- CDNA5 async DMA (global -> LDS), ASYNCcnt tracked ----------------
__device__ __forceinline__ void stage_async(uint32_t lds_addr, uint32_t goff, uint64_t gbase) {
  // GVS mode: mem = SADDR(64) + VADDR(32) ; VDST holds LDS byte address
  asm volatile("global_load_async_to_lds_b128 %0, %1, %2"
               :: "v"(lds_addr), "v"(goff), "s"(gbase)
               : "memory");
}
__device__ __forceinline__ void wait_async0() {
  asm volatile("s_wait_asynccnt 0x0" ::: "memory");
}

// One K-step of the block GEMM: A-frag from global, 4 B-frags from a fixed LDS buffer.
// All B-frags loaded into distinct registers first so the 8 ds_load_b128 can clause/overlap.
__device__ __forceinline__ void gemm_step(const h16* __restrict__ arow, int ks,
                                          const h16* __restrict__ bbuf,
                                          int lm, int ka8, int kb16,
                                          v8f& c0, v8f& c1, v8f& c2, v8f& c3) {
  v16h a  = joinv(*(const v8h*)(arow + ks + ka8),
                  *(const v8h*)(arow + ks + 16 + ka8));
  const h16* br = bbuf + lm * BROW;
  v16h b0 = joinv(*(const v8h*)(br + kb16),              *(const v8h*)(br + kb16 + 16));
  v16h b1 = joinv(*(const v8h*)(br + 16 * BROW + kb16),  *(const v8h*)(br + 16 * BROW + kb16 + 16));
  v16h b2 = joinv(*(const v8h*)(br + 32 * BROW + kb16),  *(const v8h*)(br + 32 * BROW + kb16 + 16));
  v16h b3 = joinv(*(const v8h*)(br + 48 * BROW + kb16),  *(const v8h*)(br + 48 * BROW + kb16 + 16));
  c0 = wmma_f16(a, b0, c0);
  c1 = wmma_f16(a, b1, c1);
  c2 = wmma_f16(a, b2, c2);
  c3 = wmma_f16(a, b3, c3);
}

// Block-cooperative GEMM mainloop: block = 8 waves; block tile = 128(M) x 64(N).
// Weight slice (64 rows x 32 K) staged to LDS by async DMA, software-pipelined x2
// (K/32 is even for every GEMM here: 12 or 48), so each compute site sees a
// compile-time-constant buffer pointer (no accumulator phi shuffling).
__device__ __forceinline__ void gemm_block(const h16* __restrict__ A, int lda,
                                           const h16* __restrict__ W, int ldw,
                                           int K, int m0, int n0,
                                           h16* Bsm,
                                           v8f& c0, v8f& c1, v8f& c2, v8f& c3) {
  const int tid  = threadIdx.x;
  const int lane = tid & 31;
  const int lm   = lane & 15;
  const int ka8  = (lane >> 4) * 8;
  const int kb16 = (lane >> 4) * 8;
  const h16* arow = A + (size_t)(m0 + lm) * lda;

  // staging coords: 256 threads x 16B = one 64x32 f16 tile per step
  const int srow = tid >> 2;            // 0..63  (weight row within strip)
  const int sseg = (tid & 3) * 8;       // halfs 0,8,16,24 within the 32-half K slice
  const uint64_t gbase = (uint64_t)(uintptr_t)(W + (size_t)n0 * ldw);
  const uint32_t goff0 = (uint32_t)(((size_t)srow * ldw + sseg) * 2);
  const uint32_t lds0  = (uint32_t)(uintptr_t)Bsm + (uint32_t)((srow * BROW + sseg) * 2);
  const uint32_t lds1  = lds0 + (uint32_t)(BUFH * 2);

  v8f z = {};
  c0 = z; c1 = z; c2 = z; c3 = z;

  const int nsteps = K / 32;            // even
  stage_async(lds0, goff0, gbase);      // prologue: buf0 <- ks=0
  for (int i = 0; i < nsteps; i += 2) {
    const int ks = i * 32;
    // ---- even step: compute from buf0, stage buf1 (always exists: i+1 < nsteps) ----
    wait_async0();
    __syncthreads();
    stage_async(lds1, goff0 + (uint32_t)((ks + 32) * 2), gbase);
    __builtin_prefetch(arow + ks + 128, 0, 1);           // global_prefetch_b8
    gemm_step(arow, ks, Bsm, lm, ka8, kb16, c0, c1, c2, c3);
    // ---- odd step: compute from buf1, stage buf0 for ks+64 ----
    wait_async0();
    __syncthreads();
    if (i + 2 < nsteps)
      stage_async(lds0, goff0 + (uint32_t)((ks + 64) * 2), gbase);
    gemm_step(arow, ks + 32, Bsm + BUFH, lm, ka8, kb16, c0, c1, c2, c3);
  }
}

// ---------------- kernels ----------------

__global__ __launch_bounds__(256) void conv_f32_f16(const float* __restrict__ s,
                                                    h16* __restrict__ d, int n) {
  for (int i = blockIdx.x * blockDim.x + threadIdx.x; i < n; i += gridDim.x * blockDim.x)
    d[i] = (h16)s[i];
}

// LayerNorm over channels; one thread per token (threads in a block hit consecutive l' -> coalesced)
__global__ __launch_bounds__(256) void ln_kernel(const float* __restrict__ x,
                                                 const float* __restrict__ gamma,
                                                 const float* __restrict__ beta,
                                                 h16* __restrict__ tok) {
  int t = blockIdx.x * blockDim.x + threadIdx.x;
  if (t >= KT) return;
  int b = t >> 10, l = t & 1023;
  const float* xp = x + (size_t)b * KC * KL + l;
  float s = 0.f, s2 = 0.f;
  for (int c = 0; c < KC; ++c) {
    float v = xp[(size_t)c * KL];
    s += v; s2 += v * v;
  }
  float mu = s * (1.0f / KC);
  float var = s2 * (1.0f / KC) - mu * mu;
  float rs = rsqrtf(var + 1e-5f);
  h16* out = tok + (size_t)t * KC;
  for (int c = 0; c < KC; ++c) {
    float v = xp[(size_t)c * KL];
    out[c] = (h16)((v - mu) * rs * gamma[c] + beta[c]);
  }
}

// QKV projection: tok[8192x384] x {Wq,Wk,Wv}^T -> q,k (head-major, stride 64, pad pre-zeroed), vT (dv-major)
__global__ __launch_bounds__(256) void qkv_kernel(const h16* __restrict__ tok,
                                                  const h16* __restrict__ WqH,
                                                  const h16* __restrict__ WkH,
                                                  const h16* __restrict__ WvH,
                                                  h16* __restrict__ qf,
                                                  h16* __restrict__ kf,
                                                  h16* __restrict__ vT) {
  __shared__ __align__(16) h16 Bsm[2 * BUFH];
  const int nStr = (3 * KC) / 64;                        // 18
  int ns = blockIdx.x % nStr;
  int m0 = (blockIdx.x / nStr) * 128 + (threadIdx.x >> 5) * 16;
  int n0g = ns * 64;
  int which; const h16* W; int n0;
  if (n0g < KC)            { which = 0; W = WqH; n0 = n0g; }
  else if (n0g < 2 * KC)   { which = 1; W = WkH; n0 = n0g - KC; }
  else                     { which = 2; W = WvH; n0 = n0g - 2 * KC; }
  int lane = threadIdx.x & 31;
  v8f acc[4];
  gemm_block(tok, KC, W, KC, KC, m0, n0, Bsm, acc[0], acc[1], acc[2], acc[3]);
  int lm = lane & 15, half = lane >> 4;
#pragma unroll
  for (int s = 0; s < 4; ++s) {
    int o = n0 + s * 16 + lm;
    int h = o / KDK, d = o % KDK;
#pragma unroll
    for (int r = 0; r < 8; ++r) {
      int trow = m0 + r + 8 * half;
      int b = trow >> 10, l = trow & 1023;
      size_t bh = (size_t)(b * KNH + h);
      h16 val = (h16)acc[s][r];
      if (which == 0)      qf[(bh * KL + l) * KDKP + d] = val;
      else if (which == 1) kf[(bh * KL + l) * KDKP + d] = val;
      else                 vT[(bh * KDV + d) * KL + l] = val;
    }
  }
}

// Flash attention: one wave per (b, h, 16-row Q tile); online softmax; P transposed via LDS
__global__ __launch_bounds__(32) void attn_kernel(const h16* __restrict__ qf,
                                                  const h16* __restrict__ kf,
                                                  const h16* __restrict__ vT,
                                                  const float* __restrict__ rel_bias,
                                                  const int* __restrict__ rel_idx,
                                                  h16* __restrict__ att) {
  __shared__ __align__(16) h16 Pl[16 * 32];
  int qt = blockIdx.x & 63;
  int bh = blockIdx.x >> 6;
  int h  = bh & 7;
  int lane = threadIdx.x & 31;
  int lm = lane & 15, half = lane >> 4;

  const h16* Qrow = qf + ((size_t)bh * KL + qt * 16 + lm) * KDKP;
  v16h Qa0 = load_af(Qrow, 0, lane);
  v16h Qa1 = load_af(Qrow, 32, lane);

  const float* biasrow = rel_bias + (size_t)h * KNB;
  v8f o0 = {}, o1 = {}, o2 = {};
  float mrow[8], lrow[8];
#pragma unroll
  for (int r = 0; r < 8; ++r) { mrow[r] = -3.0e38f; lrow[r] = 0.f; }

  for (int kb = 0; kb < KL; kb += 32) {
    // ---- scores: two 16x16 S tiles (keys kb..kb+31), K = 64 (padded) ----
    v8f s0 = {}, s1 = {};
    {
      const h16* krow0 = kf + ((size_t)bh * KL + kb + lm) * KDKP;
      s0 = wmma_f16(Qa0, load_bf(krow0, 0, lane), s0);
      s0 = wmma_f16(Qa1, load_bf(krow0, 32, lane), s0);
      const h16* krow1 = krow0 + (size_t)16 * KDKP;
      s1 = wmma_f16(Qa0, load_bf(krow1, 0, lane), s1);
      s1 = wmma_f16(Qa1, load_bf(krow1, 32, lane), s1);
    }
    // ---- relative bias + online softmax update ----
    float p0[8], p1[8], scale[8];
#pragma unroll
    for (int r = 0; r < 8; ++r) {
      int row = qt * 16 + r + 8 * half;
      float a0 = s0[r] + biasrow[rel_idx[row * KL + kb + lm]];
      float a1 = s1[r] + biasrow[rel_idx[row * KL + kb + 16 + lm]];
      float mt = fmaxf(a0, a1);
#pragma unroll
      for (int xm = 1; xm < 16; xm <<= 1) mt = fmaxf(mt, __shfl_xor(mt, xm, 32));
      float mn = fmaxf(mrow[r], mt);
      scale[r] = __expf(mrow[r] - mn);
      mrow[r] = mn;
      p0[r] = __expf(a0 - mn);
      p1[r] = __expf(a1 - mn);
      float rs = p0[r] + p1[r];
#pragma unroll
      for (int xm = 1; xm < 16; xm <<= 1) rs += __shfl_xor(rs, xm, 32);
      lrow[r] = lrow[r] * scale[r] + rs;
    }
#pragma unroll
    for (int r = 0; r < 8; ++r) {
      o0[r] = o0[r] * scale[r];
      o1[r] = o1[r] * scale[r];
      o2[r] = o2[r] * scale[r];
    }
    // ---- P: C-frag -> LDS -> A-frag ----
#pragma unroll
    for (int r = 0; r < 8; ++r) {
      int row = r + 8 * half;
      Pl[row * 32 + lm]      = (h16)p0[r];
      Pl[row * 32 + 16 + lm] = (h16)p1[r];
    }
    __syncthreads();
    v16h Pa;
    {
      v8h lo = *(const v8h*)(Pl + lm * 32 + half * 8);
      v8h hi = *(const v8h*)(Pl + lm * 32 + 16 + half * 8);
      Pa = joinv(lo, hi);
    }
    __syncthreads();
    // ---- O += P(16x32) * V(32x48) : 3 N-tiles from transposed V ----
    const h16* vrow = vT + ((size_t)bh * KDV + lm) * KL;
    o0 = wmma_f16(Pa, load_bf(vrow, kb, lane), o0);
    o1 = wmma_f16(Pa, load_bf(vrow + (size_t)16 * KL, kb, lane), o1);
    o2 = wmma_f16(Pa, load_bf(vrow + (size_t)32 * KL, kb, lane), o2);
  }
  // ---- normalize + store token-major f16 ----
  int b = bh >> 3;
#pragma unroll
  for (int r = 0; r < 8; ++r) {
    int t = (b << 10) + qt * 16 + r + 8 * half;
    float inv = 1.0f / lrow[r];
    h16* op = att + (size_t)t * KC + h * KDV;
    op[lm]      = (h16)(o0[r] * inv);
    op[16 + lm] = (h16)(o1[r] * inv);
    op[32 + lm] = (h16)(o2[r] * inv);
  }
}

// output projection + residual: y = x + att*Wo^T + bo ; write f32 (BCHW) and f16 (token-major)
__global__ __launch_bounds__(256) void proj_kernel(const h16* __restrict__ att,
                                                   const h16* __restrict__ WoH,
                                                   const float* __restrict__ bo,
                                                   const float* __restrict__ x,
                                                   float* __restrict__ yf32,
                                                   h16* __restrict__ yf16) {
  __shared__ __align__(16) h16 Bsm[2 * BUFH];
  const int nStr = KC / 64;                              // 6
  int ns = blockIdx.x % nStr;
  int m0 = (blockIdx.x / nStr) * 128 + (threadIdx.x >> 5) * 16;
  int n0 = ns * 64;
  int lane = threadIdx.x & 31;
  v8f acc[4];
  gemm_block(att, KC, WoH, KC, KC, m0, n0, Bsm, acc[0], acc[1], acc[2], acc[3]);
  int lm = lane & 15, half = lane >> 4;
#pragma unroll
  for (int s = 0; s < 4; ++s) {
    int n = n0 + s * 16 + lm;
    float bias = bo[n];
#pragma unroll
    for (int r = 0; r < 8; ++r) {
      int t = m0 + r + 8 * half;
      int b = t >> 10, l = t & 1023;
      size_t xi = ((size_t)b * KC + n) * KL + l;
      float v = acc[s][r] + bias + x[xi];
      yf32[xi] = v;
      yf16[(size_t)t * KC + n] = (h16)v;
    }
  }
}

// FFN up: u1 = gelu(y * fc1^T + b1)   (exact erf GELU)
__global__ __launch_bounds__(256) void ffn1_kernel(const h16* __restrict__ yf16,
                                                   const h16* __restrict__ fc1H,
                                                   const float* __restrict__ fc1b,
                                                   h16* __restrict__ u1) {
  __shared__ __align__(16) h16 Bsm[2 * BUFH];
  const int nStr = KCFF / 64;                            // 24
  int ns = blockIdx.x % nStr;
  int m0 = (blockIdx.x / nStr) * 128 + (threadIdx.x >> 5) * 16;
  int n0 = ns * 64;
  int lane = threadIdx.x & 31;
  v8f acc[4];
  gemm_block(yf16, KC, fc1H, KC, KC, m0, n0, Bsm, acc[0], acc[1], acc[2], acc[3]);
  int lm = lane & 15, half = lane >> 4;
#pragma unroll
  for (int s = 0; s < 4; ++s) {
    int n = n0 + s * 16 + lm;
    float bias = fc1b[n];
#pragma unroll
    for (int r = 0; r < 8; ++r) {
      int t = m0 + r + 8 * half;
      float u = acc[s][r] + bias;
      float g = 0.5f * u * (1.0f + erff(u * 0.70710678f));
      u1[(size_t)t * KCFF + n] = (h16)g;
    }
  }
}

// FFN down + final residual: out = y + u1 * fc2^T + b2  (BCHW f32)
__global__ __launch_bounds__(256) void ffn2_kernel(const h16* __restrict__ u1,
                                                   const h16* __restrict__ fc2H,
                                                   const float* __restrict__ fc2b,
                                                   const float* __restrict__ yf32,
                                                   float* __restrict__ out) {
  __shared__ __align__(16) h16 Bsm[2 * BUFH];
  const int nStr = KC / 64;                              // 6
  int ns = blockIdx.x % nStr;
  int m0 = (blockIdx.x / nStr) * 128 + (threadIdx.x >> 5) * 16;
  int n0 = ns * 64;
  int lane = threadIdx.x & 31;
  v8f acc[4];
  gemm_block(u1, KCFF, fc2H, KCFF, KCFF, m0, n0, Bsm, acc[0], acc[1], acc[2], acc[3]);
  int lm = lane & 15, half = lane >> 4;
#pragma unroll
  for (int s = 0; s < 4; ++s) {
    int n = n0 + s * 16 + lm;
    float bias = fc2b[n];
#pragma unroll
    for (int r = 0; r < 8; ++r) {
      int t = m0 + r + 8 * half;
      int b = t >> 10, l = t & 1023;
      size_t oi = ((size_t)b * KC + n) * KL + l;
      out[oi] = yf32[oi] + acc[s][r] + bias;
    }
  }
}

// ---------------- host launcher ----------------
extern "C" void kernel_launch(void* const* d_in, const int* in_sizes, int n_in,
                              void* d_out, int out_size, void* d_ws, size_t ws_size,
                              hipStream_t stream) {
  const float* x      = (const float*)d_in[0];
  const float* gamma  = (const float*)d_in[1];
  const float* beta   = (const float*)d_in[2];
  const float* Wq     = (const float*)d_in[3];
  const float* Wk     = (const float*)d_in[4];
  const float* Wv     = (const float*)d_in[5];
  const float* Wo     = (const float*)d_in[6];
  const float* bo     = (const float*)d_in[7];
  const float* relb   = (const float*)d_in[8];
  const float* fc1w   = (const float*)d_in[9];
  const float* fc1b   = (const float*)d_in[10];
  const float* fc2w   = (const float*)d_in[11];
  const float* fc2b   = (const float*)d_in[12];
  const int*   relidx = (const int*)d_in[13];
  float* out = (float*)d_out;

  // workspace carve (256B aligned)
  char* base = (char*)d_ws;
  size_t off = 0;
  auto alloc = [&](size_t bytes) -> void* {
    void* p = base + off;
    off = (off + bytes + 255) & ~(size_t)255;
    return p;
  };
  h16* tok  = (h16*)alloc((size_t)KT * KC * 2);
  h16* WqH  = (h16*)alloc((size_t)KC * KC * 2);
  h16* WkH  = (h16*)alloc((size_t)KC * KC * 2);
  h16* WvH  = (h16*)alloc((size_t)KC * KC * 2);
  h16* WoH  = (h16*)alloc((size_t)KC * KC * 2);
  h16* fc1H = (h16*)alloc((size_t)KCFF * KC * 2);
  h16* fc2H = (h16*)alloc((size_t)KC * KCFF * 2);
  h16* qf   = (h16*)alloc((size_t)KB * KNH * KL * KDKP * 2);
  h16* kf   = (h16*)alloc((size_t)KB * KNH * KL * KDKP * 2);
  h16* vT   = (h16*)alloc((size_t)KB * KNH * KDV * KL * 2);
  h16* att  = (h16*)alloc((size_t)KT * KC * 2);
  float* yf32 = (float*)alloc((size_t)KT * KC * 4);
  h16* yf16 = (h16*)alloc((size_t)KT * KC * 2);
  h16* u1   = (h16*)alloc((size_t)KT * KCFF * 2);
  (void)ws_size; (void)in_sizes; (void)n_in; (void)out_size;

  // weight conversions f32 -> f16
  int nw = KC * KC;
  conv_f32_f16<<<(nw + 255) / 256, 256, 0, stream>>>(Wq, WqH, nw);
  conv_f32_f16<<<(nw + 255) / 256, 256, 0, stream>>>(Wk, WkH, nw);
  conv_f32_f16<<<(nw + 255) / 256, 256, 0, stream>>>(Wv, WvH, nw);
  conv_f32_f16<<<(nw + 255) / 256, 256, 0, stream>>>(Wo, WoH, nw);
  int nf = KCFF * KC;
  conv_f32_f16<<<(nf + 255) / 256, 256, 0, stream>>>(fc1w, fc1H, nf);
  conv_f32_f16<<<(nf + 255) / 256, 256, 0, stream>>>(fc2w, fc2H, nf);

  // zero q/k so head-dim pad (48..63) contributes nothing to WMMA K=64
  hipMemsetAsync(qf, 0, (size_t)KB * KNH * KL * KDKP * 2, stream);
  hipMemsetAsync(kf, 0, (size_t)KB * KNH * KL * KDKP * 2, stream);

  // 1) LayerNorm
  ln_kernel<<<KT / 256, 256, 0, stream>>>(x, gamma, beta, tok);

  // 2) QKV projection: block tile 128x64, async-LDS staged weights
  qkv_kernel<<<(KT / 128) * ((3 * KC) / 64), 256, 0, stream>>>(tok, WqH, WkH, WvH, qf, kf, vT);

  // 3) flash attention: one wave per (b,h,qtile)
  attn_kernel<<<KB * KNH * (KL / 16), 32, 0, stream>>>(qf, kf, vT, relb, relidx, att);

  // 4) output projection + residual
  proj_kernel<<<(KT / 128) * (KC / 64), 256, 0, stream>>>(att, WoH, bo, x, yf32, yf16);

  // 5) FFN up + GELU
  ffn1_kernel<<<(KT / 128) * (KCFF / 64), 256, 0, stream>>>(yf16, fc1H, fc1b, u1);

  // 6) FFN down + final residual
  ffn2_kernel<<<(KT / 128) * (KC / 64), 256, 0, stream>>>(u1, fc2H, fc2b, yf32, out);
}
